// GCN_73684458930861
// MI455X (gfx1250) — matrix operand order repaired and checked by reading, not compiled, not run on hardware
//
#include <hip/hip_runtime.h>
#include <hip/hip_bf16.h>

typedef __bf16 bf16_t;
typedef __attribute__((ext_vector_type(8)))  __bf16 v8bf;
typedef __attribute__((ext_vector_type(16))) __bf16 v16bf;
typedef __attribute__((ext_vector_type(8)))  float  v8f;

#define N_NODES   50000
#define N_EDGES   800000
#define NUM_GRAPHS 64
#define MAX_W     512

__device__ __forceinline__ bf16_t f2bf(float f) {
  unsigned u = __builtin_bit_cast(unsigned, f);
  u += 0x7FFFu + ((u >> 16) & 1u);              // round-to-nearest-even
  unsigned short h = (unsigned short)(u >> 16);
  return __builtin_bit_cast(bf16_t, h);
}

// Relaxed agent-scope f32 atomic add -> native global_atomic_add_f32 (no return)
__device__ __forceinline__ void afadd(float* p, float v) {
  (void)__hip_atomic_fetch_add(p, v, __ATOMIC_RELAXED, __HIP_MEMORY_SCOPE_AGENT);
}

// ---------------- graph-norm precompute ----------------
__global__ void k_deg_init(float* __restrict__ deg) {
  int i = blockIdx.x * blockDim.x + threadIdx.x;
  if (i < N_NODES) deg[i] = 1.0f;               // self loop
}
__global__ void k_deg_edges(const long long* __restrict__ dst, float* __restrict__ deg) {
  int e = blockIdx.x * blockDim.x + threadIdx.x;
  if (e < N_EDGES) afadd(&deg[(int)dst[e]], 1.0f);
}
__global__ void k_dinv(const float* __restrict__ deg, float* __restrict__ dinv) {
  int i = blockIdx.x * blockDim.x + threadIdx.x;
  if (i < N_NODES) dinv[i] = rsqrtf(deg[i]);    // deg >= 1 always
}
__global__ void k_enorm(const long long* __restrict__ src, const long long* __restrict__ dst,
                        const float* __restrict__ dinv, float* __restrict__ enorm) {
  int e = blockIdx.x * blockDim.x + threadIdx.x;
  if (e < N_EDGES) enorm[e] = dinv[(int)src[e]] * dinv[(int)dst[e]];
}

// ---------------- casts / weight transpose ----------------
__global__ void k_cast_x(const float* __restrict__ x, bf16_t* __restrict__ hbf, int count) {
  int i = blockIdx.x * blockDim.x + threadIdx.x;
  if (i < count) hbf[i] = f2bf(x[i]);
}
// W: [K, N] row-major f32  ->  WT: [N, K] row-major bf16
__global__ void k_wt(const float* __restrict__ W, bf16_t* __restrict__ WT, int K, int N) {
  int i = blockIdx.x * blockDim.x + threadIdx.x;
  if (i >= K * N) return;
  int k = i / N, n = i - k * N;
  WT[(size_t)n * K + k] = f2bf(W[i]);
}

// ---------------- bf16 WMMA GEMM + self-loop epilogue ----------------
// tmp[M,N] = A[M,K] * WT[N,K]^T ; agg[M,N] = dinv[m]^2 * tmp[M,N]
__global__ void __launch_bounds__(256)
k_gemm_bf16_wmma(const bf16_t* __restrict__ A, const bf16_t* __restrict__ BT,
                 float* __restrict__ tmp, float* __restrict__ agg,
                 const float* __restrict__ dinv,
                 int K, int N, int totalTiles, int ntilesN) {
  int wave = threadIdx.x >> 5;
  int lane = threadIdx.x & 31;
  int tile = blockIdx.x * 8 + wave;
  if (tile >= totalTiles) return;               // wave-uniform: EXEC all-ones for WMMA
  int mt = tile / ntilesN;
  int nt = tile - mt * ntilesN;
  int r    = lane & 15;
  int half = lane >> 4;
  const bf16_t* arow = A  + (size_t)(mt * 16 + r) * K;
  const bf16_t* brow = BT + (size_t)(nt * 16 + r) * K;
  v8f acc = {};
  for (int k0 = 0; k0 < K; k0 += 32) {
    // A 16x32 bf16 fragment: lane holds row r, K = {h*8..h*8+7} U {16+h*8..16+h*8+7}
    v8bf a0 = *(const v8bf*)(arow + k0 + half * 8);
    v8bf a1 = *(const v8bf*)(arow + k0 + 16 + half * 8);
    // B 32x16 bf16 fragment: lane holds col r, K = {h*8..h*8+7} U {16+h*8..16+h*8+7}
    v8bf b0 = *(const v8bf*)(brow + k0 + half * 8);
    v8bf b1 = *(const v8bf*)(brow + k0 + 16 + half * 8);
    v16bf av = __builtin_shufflevector(a0, a1, 0,1,2,3,4,5,6,7,8,9,10,11,12,13,14,15);
    v16bf bv = __builtin_shufflevector(b0, b1, 0,1,2,3,4,5,6,7,8,9,10,11,12,13,14,15);
    acc = __builtin_amdgcn_wmma_f32_16x16x32_bf16(false, av, false, bv,
                                                  (short)0, acc, false, false);
  }
  // C/D layout: VGPR i, lane l -> row i + 8*(l>>4), col l&15
  int rowBase = mt * 16 + half * 8;
  size_t off = (size_t)rowBase * N + nt * 16 + r;
#pragma unroll
  for (int i = 0; i < 8; ++i) {
    float v = acc[i];
    float d = dinv[rowBase + i];
    tmp[off + (size_t)i * N] = v;
    agg[off + (size_t)i * N] = d * d * v;       // self-loop term, also inits accumulator
  }
}

// ---------------- edge scatter: agg[dst] += norm[e] * tmp[src] ----------------
__global__ void __launch_bounds__(256)
k_scatter(const float* __restrict__ tmp, const long long* __restrict__ src,
          const long long* __restrict__ dst, const float* __restrict__ enorm,
          float* __restrict__ agg, int F) {
  int fq  = F >> 2;           // float4 chunks per row (16..128), divides 256
  int epb = 256 / fq;
  int e = blockIdx.x * epb + threadIdx.x / fq;
  if (e >= N_EDGES) return;
  int q = threadIdx.x % fq;
  int s = (int)src[e];
  int d = (int)dst[e];
  float w = enorm[e];
  float4 v = *(const float4*)(tmp + (size_t)s * F + q * 4);
  float* ap = agg + (size_t)d * F + q * 4;
  afadd(ap + 0, w * v.x);
  afadd(ap + 1, w * v.y);
  afadd(ap + 2, w * v.z);
  afadd(ap + 3, w * v.w);
}

// h = relu(agg + b); bf16 always (next GEMM), f32 only when needed (pooling)
__global__ void k_bias_relu(const float* __restrict__ agg, const float* __restrict__ b,
                            bf16_t* __restrict__ hbf, float* __restrict__ hf32,
                            int F, int writeF32) {
  int i = blockIdx.x * blockDim.x + threadIdx.x;
  if (i >= N_NODES * F) return;
  int f = i % F;
  float v = agg[i] + b[f];
  v = fmaxf(v, 0.0f);
  hbf[i] = f2bf(v);
  if (writeF32) hf32[i] = v;
}

// ---------------- pooling + output ----------------
__global__ void k_pool_zero(float* __restrict__ sums, float* __restrict__ cnts) {
  int i = blockIdx.x * blockDim.x + threadIdx.x;
  if (i < NUM_GRAPHS * 64) sums[i] = 0.0f;
  if (i < NUM_GRAPHS)      cnts[i] = 0.0f;
}
__global__ void k_pool(const float* __restrict__ h, const long long* __restrict__ batch,
                       float* __restrict__ sums, float* __restrict__ cnts) {
  int i = blockIdx.x * blockDim.x + threadIdx.x;
  if (i >= N_NODES * 16) return;
  int n = i >> 4, q = i & 15;
  int g = (int)batch[n];
  float4 v = *(const float4*)(h + (size_t)n * 64 + q * 4);
  float* sp = sums + g * 64 + q * 4;
  afadd(sp + 0, v.x); afadd(sp + 1, v.y);
  afadd(sp + 2, v.z); afadd(sp + 3, v.w);
  if (q == 0) afadd(&cnts[g], 1.0f);
}
__global__ void k_final(const float* __restrict__ sums, const float* __restrict__ cnts,
                        const float* __restrict__ Wout, const float* __restrict__ bout,
                        float* __restrict__ out) {
  int g = threadIdx.x;
  if (g >= NUM_GRAPHS) return;
  float c = fmaxf(cnts[g], 1.0f);
  float acc = bout[0];
  for (int f = 0; f < 64; ++f) acc += (sums[g * 64 + f] / c) * Wout[f];
  out[g] = acc;
}

static inline int gb(long long n, int t) { return (int)((n + t - 1) / t); }

extern "C" void kernel_launch(void* const* d_in, const int* in_sizes, int n_in,
                              void* d_out, int out_size, void* d_ws, size_t ws_size,
                              hipStream_t stream) {
  const float*     x     = (const float*)d_in[0];
  const long long* ei    = (const long long*)d_in[1];   // int64 [2, E]
  const long long* batch = (const long long*)d_in[2];   // int64 [N]
  const float* Ws[7]; const float* bs[7];
  for (int i = 0; i < 7; ++i) { Ws[i] = (const float*)d_in[3 + i]; bs[i] = (const float*)d_in[10 + i]; }
  const float* Wout = (const float*)d_in[17];
  const float* bout = (const float*)d_in[18];
  float* out = (float*)d_out;

  const long long* src = ei;
  const long long* dst = ei + N_EDGES;

  // workspace carve-out (256B aligned)
  char* p = (char*)d_ws;
  auto alloc = [&](size_t bytes) -> void* {
    void* q = (void*)p; p += (bytes + 255) & ~(size_t)255; return q;
  };
  float*  deg   = (float*) alloc((size_t)N_NODES * 4);
  float*  dinv  = (float*) alloc((size_t)N_NODES * 4);
  float*  enorm = (float*) alloc((size_t)N_EDGES * 4);
  bf16_t* Hbf   = (bf16_t*)alloc((size_t)N_NODES * MAX_W * 2);
  float*  tmp   = (float*) alloc((size_t)N_NODES * MAX_W * 4);  // GEMM out, reused as hf32
  float*  agg   = (float*) alloc((size_t)N_NODES * MAX_W * 4);
  bf16_t* WT    = (bf16_t*)alloc((size_t)MAX_W * MAX_W * 2);
  float*  sums  = (float*) alloc((size_t)NUM_GRAPHS * 64 * 4);
  float*  cnts  = (float*) alloc((size_t)NUM_GRAPHS * 4);
  float*  hf32  = tmp;   // bias_relu runs after scatter; tmp is dead by then

  // ---- normalization precompute ----
  k_deg_init <<<gb(N_NODES, 256), 256, 0, stream>>>(deg);
  k_deg_edges<<<gb(N_EDGES, 256), 256, 0, stream>>>(dst, deg);
  k_dinv     <<<gb(N_NODES, 256), 256, 0, stream>>>(deg, dinv);
  k_enorm    <<<gb(N_EDGES, 256), 256, 0, stream>>>(src, dst, dinv, enorm);

  // ---- input cast ----
  k_cast_x<<<gb((long long)N_NODES * 128, 256), 256, 0, stream>>>(x, Hbf, N_NODES * 128);

  const int fins [7] = {128, 64, 128, 256, 512, 256, 128};
  const int fouts[7] = { 64, 128, 256, 512, 256, 128,  64};

  for (int l = 0; l < 7; ++l) {
    int K = fins[l], F = fouts[l];
    // weight transpose+cast
    k_wt<<<gb((long long)K * F, 256), 256, 0, stream>>>(Ws[l], WT, K, F);
    // tmp = Hbf @ W (bf16 WMMA, f32 accum); agg = dinv^2 * tmp (fused epilogue)
    int ntilesN = F >> 4;
    int totalTiles = (N_NODES >> 4) * ntilesN;
    k_gemm_bf16_wmma<<<gb(totalTiles, 8), 256, 0, stream>>>(Hbf, WT, tmp, agg, dinv,
                                                            K, F, totalTiles, ntilesN);
    // edge scatter (hardware f32 atomics)
    int epb = 256 / (F >> 2);
    k_scatter<<<gb(N_EDGES, epb), 256, 0, stream>>>(tmp, src, dst, enorm, agg, F);
    // bias + relu -> bf16 (next layer); f32 copy only for last layer (pooling)
    k_bias_relu<<<gb((long long)N_NODES * F, 256), 256, 0, stream>>>(agg, bs[l], Hbf, hf32,
                                                                     F, (l == 6) ? 1 : 0);
  }

  // ---- global mean pool (+ output projection) ----
  k_pool_zero<<<gb(NUM_GRAPHS * 64, 256), 256, 0, stream>>>(sums, cnts);
  k_pool     <<<gb((long long)N_NODES * 16, 256), 256, 0, stream>>>(hf32, batch, sums, cnts);
  k_final    <<<1, 64, 0, stream>>>(sums, cnts, Wout, bout, out);
}
